// WireframeDetector_10943576671014
// MI455X (gfx1250) — matrix operand order repaired and checked by reading, not compiled, not run on hardware
//
#include <hip/hip_runtime.h>
#include <cstdint>

// ---------------------------------------------------------------------------
// Types for CDNA5 WMMA
// ---------------------------------------------------------------------------
typedef __attribute__((ext_vector_type(16))) __bf16 v16bf;
typedef __attribute__((ext_vector_type(8)))  float  v8f;

union FragAB {
    v16bf v;
    uint4 q[2];
};

__device__ __forceinline__ __bf16 f2bf_hi(float x) { return (__bf16)x; }
__device__ __forceinline__ __bf16 f2bf_lo(float x, __bf16 hi) { return (__bf16)(x - (float)hi); }

// ---------------------------------------------------------------------------
// Kernel 1: 3x3 NMS on jloc (128x128), SAME padding with -inf identity
// ---------------------------------------------------------------------------
__global__ void wf_nms_kernel(const float* __restrict__ jloc, float* __restrict__ nms) {
    int i = blockIdx.x * blockDim.x + threadIdx.x;
    if (i >= 16384) return;
    int y = i >> 7, x = i & 127;
    float v = jloc[i];
    float m = -3.4e38f;
    #pragma unroll
    for (int dy = -1; dy <= 1; ++dy) {
        int yy = y + dy;
        if (yy < 0 || yy > 127) continue;
        #pragma unroll
        for (int dx = -1; dx <= 1; ++dx) {
            int xx = x + dx;
            if (xx < 0 || xx > 127) continue;
            m = fmaxf(m, jloc[yy * 128 + xx]);
        }
    }
    nms[i] = (v == m) ? v : 0.0f;
}

// ---------------------------------------------------------------------------
// Kernel 2: iterative top-300 (descending, lowest-index tie break like
// jax.lax.top_k) + junction coordinate computation.
// ---------------------------------------------------------------------------
__global__ void wf_topk_junc_kernel(const float* __restrict__ nms,
                                    const float* __restrict__ joff,
                                    float* __restrict__ jx, float* __restrict__ jy) {
    __shared__ unsigned removed[512];      // 16384 bits
    __shared__ float rv[256];
    __shared__ int   ri[256];
    __shared__ int   sIdx[300];
    const int tid = threadIdx.x;

    removed[tid] = 0u;
    removed[tid + 256] = 0u;
    __syncthreads();

    for (int it = 0; it < 300; ++it) {
        float best = -3.4e38f;
        int bi = 1 << 30;
        #pragma unroll 4
        for (int e = 0; e < 64; ++e) {
            int i = tid * 64 + e;
            if (!(removed[i >> 5] & (1u << (i & 31)))) {
                float v = nms[i];
                if (v > best) { best = v; bi = i; }   // ascending scan -> lowest idx wins ties
            }
        }
        rv[tid] = best; ri[tid] = bi;
        __syncthreads();
        for (int s = 128; s > 0; s >>= 1) {
            if (tid < s) {
                float ov = rv[tid + s]; int oi = ri[tid + s];
                if (ov > rv[tid] || (ov == rv[tid] && oi < ri[tid])) { rv[tid] = ov; ri[tid] = oi; }
            }
            __syncthreads();
        }
        if (tid == 0) {
            int w = ri[0];
            sIdx[it] = w;
            removed[w >> 5] |= (1u << (w & 31));
        }
        __syncthreads();
    }

    // juncs: x = idx%128 + sigmoid(joff0[idx]);  y = idx/128 + sigmoid(joff1[idx])
    for (int j = tid; j < 300; j += 256) {
        int id = sIdx[j];
        float sx = 1.0f / (1.0f + __expf(-joff[id]));
        float sy = 1.0f / (1.0f + __expf(-joff[16384 + id]));
        jx[j] = (float)(id & 127) + sx;
        jy[j] = (float)(id >> 7) + sy;
    }
}

// ---------------------------------------------------------------------------
// Kernel 3: transpose + hi/lo bf16 split of a 1024x1024 fp32 weight (K-major
// in, N-major out) so GEMM B-fragments are K-contiguous per lane.
// ---------------------------------------------------------------------------
__global__ void wf_wsplit_kernel(const float* __restrict__ W,
                                 __bf16* __restrict__ th, __bf16* __restrict__ tl) {
    int i = blockIdx.x * blockDim.x + threadIdx.x;    // i = k*1024 + n
    if (i >= 1024 * 1024) return;
    int k = i >> 10, n = i & 1023;
    float v = W[i];
    __bf16 hb = f2bf_hi(v);
    th[n * 1024 + k] = hb;
    tl[n * 1024 + k] = f2bf_lo(v, hb);
}

// ---------------------------------------------------------------------------
// Kernel 4: per-line bilinear sampling + max-pool(4) -> feats hi/lo bf16.
// ---------------------------------------------------------------------------
__global__ __launch_bounds__(128)
void wf_sample_kernel(const float* __restrict__ loi, const int* __restrict__ edge,
                      const float* __restrict__ jx, const float* __restrict__ jy,
                      __bf16* __restrict__ fh, __bf16* __restrict__ fl) {
    __shared__ int   so00[32], so10[32], so01[32], so11[32];
    __shared__ float sw00[32], sw10[32], sw01[32], sw11[32];
    const int line = blockIdx.x;
    const int tid = threadIdx.x;

    if (tid < 32) {
        int e0 = edge[line * 2 + 0];
        int e1 = edge[line * 2 + 1];
        float t = (float)tid * (1.0f / 31.0f);
        float px = jx[e0] * t + jx[e1] * (1.0f - t) - 0.5f;
        float py = jy[e0] * t + jy[e1] * (1.0f - t) - 0.5f;
        float px0 = fminf(fmaxf(floorf(px), 0.0f), 127.0f);
        float py0 = fminf(fmaxf(floorf(py), 0.0f), 127.0f);
        float px1 = fminf(px0 + 1.0f, 127.0f);
        float py1 = fminf(py0 + 1.0f, 127.0f);
        int x0 = (int)px0, y0 = (int)py0, x1 = (int)px1, y1 = (int)py1;
        so00[tid] = y0 * 128 + x0;
        so10[tid] = y1 * 128 + x0;
        so01[tid] = y0 * 128 + x1;
        so11[tid] = y1 * 128 + x1;
        sw00[tid] = (py1 - py) * (px1 - px);
        sw10[tid] = (py - py0) * (px1 - px);
        sw01[tid] = (py1 - py) * (px - px0);
        sw11[tid] = (py - py0) * (px - px0);
    }
    __syncthreads();

    const int c = tid;                       // channel
    const float* base = loi + c * 16384;
    #pragma unroll
    for (int j = 0; j < 8; ++j) {
        float m = -3.4e38f;
        #pragma unroll
        for (int q = 0; q < 4; ++q) {
            int p = j * 4 + q;
            float v = base[so00[p]] * sw00[p] + base[so10[p]] * sw10[p]
                    + base[so01[p]] * sw01[p] + base[so11[p]] * sw11[p];
            m = fmaxf(m, v);
        }
        __bf16 hb = f2bf_hi(m);
        int o = line * 1024 + c * 8 + j;
        fh[o] = hb;
        fl[o] = f2bf_lo(m, hb);
    }
}

// ---------------------------------------------------------------------------
// Kernel 5: split-bf16 GEMM  C = relu(A @ B(as [N][K]) + bias), fp32-accurate
// via hi*hi + hi*lo + lo*hi.  LDS double-buffered pipeline: global loads for
// tile k+1 overlap the WMMAs of tile k.  Out-of-range A rows are handled by
// clamped addressing (row m only feeds output row m, which the epilogue
// discards), so the staging path is branch/exec-mask free.
// Block tile 128x128, 8 waves, wave tile 32x64 (2x4 accumulators).
// ---------------------------------------------------------------------------
__global__ __launch_bounds__(256)
void wf_gemm_kernel(const __bf16* __restrict__ Ah, const __bf16* __restrict__ Al,
                    const __bf16* __restrict__ Bh, const __bf16* __restrict__ Bl,
                    const float* __restrict__ bias,
                    __bf16* __restrict__ Oh, __bf16* __restrict__ Ol, int M) {
    __shared__ alignas(16) __bf16 sAh[2][128 * 32];
    __shared__ alignas(16) __bf16 sAl[2][128 * 32];
    __shared__ alignas(16) __bf16 sBh[2][128 * 32];
    __shared__ alignas(16) __bf16 sBl[2][128 * 32];

    const int tid  = threadIdx.x;
    const int lane = tid & 31;
    const int wave = tid >> 5;
    const int wm = wave & 3;          // 0..3  -> 32-row strip
    const int wn = wave >> 2;         // 0..1  -> 64-col strip
    const int hlf = lane >> 4;        // lane half
    const int l16 = lane & 15;
    const int tileM = blockIdx.x * 128;
    const int tileN = blockIdx.y * 128;

    // per-thread staging geometry (loop invariant): 2 uint4 chunks per array
    const int row0 = (tid * 2 + 0) >> 2, kc0 = ((tid * 2 + 0) & 3) * 8;
    const int row1 = (tid * 2 + 1) >> 2, kc1 = ((tid * 2 + 1) & 3) * 8;
    // clamp OOB A rows to a valid row: row m only contributes to output row m,
    // and the epilogue never stores rows >= M, so duplicated data is harmless.
    const int rA0 = min(tileM + row0, M - 1);
    const int rA1 = min(tileM + row1, M - 1);
    const __bf16* gA0h = Ah + (size_t)rA0 * 1024 + kc0;
    const __bf16* gA0l = Al + (size_t)rA0 * 1024 + kc0;
    const __bf16* gA1h = Ah + (size_t)rA1 * 1024 + kc1;
    const __bf16* gA1l = Al + (size_t)rA1 * 1024 + kc1;
    const __bf16* gB0h = Bh + (size_t)(tileN + row0) * 1024 + kc0;
    const __bf16* gB0l = Bl + (size_t)(tileN + row0) * 1024 + kc0;
    const __bf16* gB1h = Bh + (size_t)(tileN + row1) * 1024 + kc1;
    const __bf16* gB1l = Bl + (size_t)(tileN + row1) * 1024 + kc1;

    v8f acc[2][4];
    #pragma unroll
    for (int i = 0; i < 2; ++i)
        #pragma unroll
        for (int j = 0; j < 4; ++j) { v8f z = {}; acc[i][j] = z; }

    // stage tile at K-offset k0 into LDS buffer `buf` (transient registers,
    // 8 unconditional global_load_b128 + 8 ds_store_b128)
    auto stage = [&](int k0, int buf) {
        uint4 r0 = *(const uint4*)(gA0h + k0);
        uint4 r1 = *(const uint4*)(gA0l + k0);
        uint4 r2 = *(const uint4*)(gB0h + k0);
        uint4 r3 = *(const uint4*)(gB0l + k0);
        uint4 r4 = *(const uint4*)(gA1h + k0);
        uint4 r5 = *(const uint4*)(gA1l + k0);
        uint4 r6 = *(const uint4*)(gB1h + k0);
        uint4 r7 = *(const uint4*)(gB1l + k0);
        *(uint4*)(&sAh[buf][row0 * 32 + kc0]) = r0;
        *(uint4*)(&sAl[buf][row0 * 32 + kc0]) = r1;
        *(uint4*)(&sBh[buf][row0 * 32 + kc0]) = r2;
        *(uint4*)(&sBl[buf][row0 * 32 + kc0]) = r3;
        *(uint4*)(&sAh[buf][row1 * 32 + kc1]) = r4;
        *(uint4*)(&sAl[buf][row1 * 32 + kc1]) = r5;
        *(uint4*)(&sBh[buf][row1 * 32 + kc1]) = r6;
        *(uint4*)(&sBl[buf][row1 * 32 + kc1]) = r7;
    };

    stage(0, 0);
    __syncthreads();

    for (int kt = 0; kt < 32; ++kt) {
        const int cur = kt & 1;

        // stage next tile into the other buffer (overlaps WMMAs below);
        // safe with a single barrier: buffer cur^1 was last read at kt-1.
        if (kt < 31) stage((kt + 1) * 32, cur ^ 1);
        if (kt + 2 < 32) {
            int k2 = (kt + 2) * 32;
            __builtin_prefetch(gA0h + k2, 0, 1);
            __builtin_prefetch(gB0h + k2, 0, 1);
        }

        // ---- A fragments (ISA 16-bit 16x32 lane layout), held across block ----
        FragAB fa_h[2], fa_l[2];
        #pragma unroll
        for (int mt = 0; mt < 2; ++mt) {
            int r = wm * 32 + mt * 16 + l16;     // A row held by this lane
            const __bf16* ph = &sAh[cur][r * 32 + hlf * 8];
            fa_h[mt].q[0] = *(const uint4*)(ph);        // k = hlf*8 + 0..7
            fa_h[mt].q[1] = *(const uint4*)(ph + 16);   // k = hlf*8 + 16..23
            const __bf16* pl = &sAl[cur][r * 32 + hlf * 8];
            fa_l[mt].q[0] = *(const uint4*)(pl);
            fa_l[mt].q[1] = *(const uint4*)(pl + 16);
        }

        // ---- B fragments per nt (transient), 3-term split-bf16 MMA ----
        #pragma unroll
        for (int nt = 0; nt < 4; ++nt) {
            int n = wn * 64 + nt * 16 + l16;     // B column held by this lane
            FragAB fb_h, fb_l;
            const __bf16* ph = &sBh[cur][n * 32 + hlf * 8];   // k = hlf*8 + 0..7, hlf*8 + 16..23
            fb_h.q[0] = ((const uint4*)ph)[0];
            fb_h.q[1] = ((const uint4*)ph)[2];
            const __bf16* pl = &sBl[cur][n * 32 + hlf * 8];
            fb_l.q[0] = ((const uint4*)pl)[0];
            fb_l.q[1] = ((const uint4*)pl)[2];
            #pragma unroll
            for (int mt = 0; mt < 2; ++mt) {
                acc[mt][nt] = __builtin_amdgcn_wmma_f32_16x16x32_bf16(
                    false, fa_l[mt].v, false, fb_h.v, (short)0, acc[mt][nt], false, false);
                acc[mt][nt] = __builtin_amdgcn_wmma_f32_16x16x32_bf16(
                    false, fa_h[mt].v, false, fb_l.v, (short)0, acc[mt][nt], false, false);
                acc[mt][nt] = __builtin_amdgcn_wmma_f32_16x16x32_bf16(
                    false, fa_h[mt].v, false, fb_h.v, (short)0, acc[mt][nt], false, false);
            }
        }
        __syncthreads();
    }

    // ---- epilogue: bias + relu, re-split hi/lo bf16 ----
    #pragma unroll
    for (int mt = 0; mt < 2; ++mt)
        #pragma unroll
        for (int nt = 0; nt < 4; ++nt) {
            int col = tileN + wn * 64 + nt * 16 + l16;
            float bv = bias[col];
            #pragma unroll
            for (int r = 0; r < 8; ++r) {
                int rowg = tileM + wm * 32 + mt * 16 + hlf * 8 + r;
                if (rowg < M) {
                    float v = acc[mt][nt][r] + bv;
                    v = v > 0.0f ? v : 0.0f;
                    __bf16 hb = f2bf_hi(v);
                    Oh[rowg * 1024 + col] = hb;
                    Ol[rowg * 1024 + col] = f2bf_lo(v, hb);
                }
            }
        }
}

// ---------------------------------------------------------------------------
// Kernel 6: logits = h2 @ W3 + b3   (1024 x 3, wave-per-line dot products)
// ---------------------------------------------------------------------------
__global__ __launch_bounds__(256)
void wf_final_kernel(const __bf16* __restrict__ h2h, const __bf16* __restrict__ h2l,
                     const float* __restrict__ W3, const float* __restrict__ b3,
                     float* __restrict__ out, int M) {
    const int wave = threadIdx.x >> 5;
    const int lane = threadIdx.x & 31;
    const int line = blockIdx.x * 8 + wave;
    if (line >= M) return;
    const __bf16* ph = h2h + line * 1024;
    const __bf16* pl = h2l + line * 1024;
    float a0 = 0.f, a1 = 0.f, a2 = 0.f;
    for (int k = lane; k < 1024; k += 32) {
        float x = (float)ph[k] + (float)pl[k];
        a0 += x * W3[k * 3 + 0];
        a1 += x * W3[k * 3 + 1];
        a2 += x * W3[k * 3 + 2];
    }
    #pragma unroll
    for (int o = 16; o > 0; o >>= 1) {
        a0 += __shfl_xor(a0, o, 32);
        a1 += __shfl_xor(a1, o, 32);
        a2 += __shfl_xor(a2, o, 32);
    }
    if (lane == 0) {
        out[line * 3 + 0] = a0 + b3[0];
        out[line * 3 + 1] = a1 + b3[1];
        out[line * 3 + 2] = a2 + b3[2];
    }
}

// ---------------------------------------------------------------------------
// Launch
// ---------------------------------------------------------------------------
extern "C" void kernel_launch(void* const* d_in, const int* in_sizes, int n_in,
                              void* d_out, int out_size, void* d_ws, size_t ws_size,
                              hipStream_t stream) {
    const float* jloc = (const float*)d_in[0];
    const float* joff = (const float*)d_in[1];
    const float* loi  = (const float*)d_in[2];
    const int*   edge = (const int*)d_in[3];
    const float* W1   = (const float*)d_in[4];
    const float* b1   = (const float*)d_in[5];
    const float* W2   = (const float*)d_in[6];
    const float* b2   = (const float*)d_in[7];
    const float* W3   = (const float*)d_in[8];
    const float* b3   = (const float*)d_in[9];
    float* out = (float*)d_out;

    const int M = 20000;
    char* ws = (char*)d_ws;
    size_t off = 0;
    auto alloc = [&](size_t bytes) -> void* {
        void* p = ws + off;
        off = (off + bytes + 255) & ~(size_t)255;
        return p;
    };

    float*  nms  = (float*)alloc(16384 * sizeof(float));
    float*  jx   = (float*)alloc(300 * sizeof(float));
    float*  jy   = (float*)alloc(300 * sizeof(float));
    __bf16* fh   = (__bf16*)alloc((size_t)M * 1024 * sizeof(__bf16));   // feats hi / later h2 hi
    __bf16* fl   = (__bf16*)alloc((size_t)M * 1024 * sizeof(__bf16));   // feats lo / later h2 lo
    __bf16* h1h  = (__bf16*)alloc((size_t)M * 1024 * sizeof(__bf16));
    __bf16* h1l  = (__bf16*)alloc((size_t)M * 1024 * sizeof(__bf16));
    __bf16* w1th = (__bf16*)alloc((size_t)1024 * 1024 * sizeof(__bf16));
    __bf16* w1tl = (__bf16*)alloc((size_t)1024 * 1024 * sizeof(__bf16));
    __bf16* w2th = (__bf16*)alloc((size_t)1024 * 1024 * sizeof(__bf16));
    __bf16* w2tl = (__bf16*)alloc((size_t)1024 * 1024 * sizeof(__bf16));

    wf_nms_kernel<<<64, 256, 0, stream>>>(jloc, nms);
    wf_topk_junc_kernel<<<1, 256, 0, stream>>>(nms, joff, jx, jy);
    wf_wsplit_kernel<<<4096, 256, 0, stream>>>(W1, w1th, w1tl);
    wf_wsplit_kernel<<<4096, 256, 0, stream>>>(W2, w2th, w2tl);
    wf_sample_kernel<<<M, 128, 0, stream>>>(loi, edge, jx, jy, fh, fl);

    dim3 ggrid((M + 127) / 128, 8);
    wf_gemm_kernel<<<ggrid, 256, 0, stream>>>(fh, fl, w1th, w1tl, b1, h1h, h1l, M);
    wf_gemm_kernel<<<ggrid, 256, 0, stream>>>(h1h, h1l, w2th, w2tl, b2, fh, fl, M);  // h2 -> feats bufs
    wf_final_kernel<<<(M + 7) / 8, 256, 0, stream>>>(fh, fl, W3, b3, out, M);
}
